// MOELoraLinear_20469814133119
// MI455X (gfx1250) — matrix operand order repaired and checked by reading, not compiled, not run on hardware
//
#include <hip/hip_runtime.h>

typedef __bf16 bf16_t;
typedef __attribute__((ext_vector_type(16))) __bf16 v16bf;
typedef __attribute__((ext_vector_type(2)))  __bf16 bf16x2;
typedef __attribute__((ext_vector_type(8)))  float  v8f;
typedef __attribute__((ext_vector_type(2)))  float  f32x2;

static constexpr float SCALING = 2.0f;

static constexpr int Bb = 4, Ss = 2048, Dd = 4096, Kk = 8, Rr = 16;
static constexpr int M_TOTAL = Bb * Ss;   // 8192
static constexpr int KR      = Kk * Rr;   // 128

// GEMM tiling
static constexpr int BM = 128, BN = 128, BK = 64;
static constexpr int LDT = BK + 8;        // padded LDS row stride (bf16) = 72
static constexpr int RBLK = 32;           // router partial blocks per batch

// ---------------------------------------------------------------------------
// Packed fp32 pair -> bf16 pair (one v_cvt_pk_bf16_f32).
// ---------------------------------------------------------------------------
__device__ __forceinline__ unsigned cvt2(float a, float b) {
  f32x2 f = {a, b};
  bf16x2 h = __builtin_convertvector(f, bf16x2);
  union { bf16x2 h; unsigned u; } cv;
  cv.h = h;
  return cv.u;
}

// Load 32 fp32 and convert to 32 bf16 packed in 4 uint4 (register staging).
__device__ __forceinline__ void load_cvt32(const float* __restrict__ p, uint4 out[4]) {
#pragma unroll
  for (int i = 0; i < 4; ++i) {
    float4 f0 = ((const float4*)p)[2 * i];
    float4 f1 = ((const float4*)p)[2 * i + 1];
    out[i].x = cvt2(f0.x, f0.y);
    out[i].y = cvt2(f0.z, f0.w);
    out[i].z = cvt2(f1.x, f1.y);
    out[i].w = cvt2(f1.z, f1.w);
  }
}

__device__ __forceinline__ void load_bf32(const bf16_t* __restrict__ p, uint4 out[4]) {
  const uint4* q = (const uint4*)p;
#pragma unroll
  for (int i = 0; i < 4; ++i) out[i] = q[i];
}

__device__ __forceinline__ void publish(bf16_t* lds, int srow, int skh, const uint4 r[4]) {
  uint4* d = (uint4*)&lds[srow * LDT + skh];
#pragma unroll
  for (int i = 0; i < 4; ++i) d[i] = r[i];
}

// ---------------------------------------------------------------------------
// Fragment loads from LDS (wave32).
// A 16x32 bf16 layout (ISA 7.12.2): lanes 0-15 hold K=0..7 then 16..23,
// lanes 16-31 hold K=8..15 then 24..31, row M = lane%16.
// ---------------------------------------------------------------------------
__device__ __forceinline__ v16bf load_a_frag(const bf16_t* lds, int mrow, int ks) {
  const int lane = threadIdx.x & 31;
  const int r = lane & 15, h = lane >> 4;
  const int base = (mrow + r) * LDT + ks * 32;
  union { uint4 u[2]; v16bf v; } t;
  t.u[0] = *(const uint4*)&lds[base + 8 * h];
  t.u[1] = *(const uint4*)&lds[base + 16 + 8 * h];
  return t.v;
}

// B 32x16 bf16: lanes 0-15 hold K=0..7 then 16..23, lanes 16-31 hold K=8..15 then 24..31, col N = lane%16.
__device__ __forceinline__ v16bf load_b_frag(const bf16_t* lds, int nrow, int ks) {
  const int lane = threadIdx.x & 31;
  const int c = lane & 15, h = lane >> 4;
  const uint4* p = (const uint4*)&lds[(nrow + c) * LDT + ks * 32 + 8 * h];
  union { uint4 u[2]; v16bf v; } t;
  t.u[0] = p[0];
  t.u[1] = p[2];
  return t.v;
}

__device__ __forceinline__ v8f wmma_bf16(v16bf a, v16bf b, v8f c) {
  return __builtin_amdgcn_wmma_f32_16x16x32_bf16(false, a, false, b, (short)0, c,
                                                 false, false);
}

// ---------------------------------------------------------------------------
// Router pass 1: partial logits over S-slices (deterministic 2-pass reduction)
// grid = Bb*RBLK blocks of 256 threads; part[b][blk][k]
// ---------------------------------------------------------------------------
__global__ __launch_bounds__(256) void router_partial(const float* __restrict__ x,
                                                      const float* __restrict__ router_w,
                                                      float* __restrict__ part) {
  const int b   = blockIdx.x / RBLK;
  const int blk = blockIdx.x % RBLK;
  const int srows = Ss / RBLK;  // 64
  float acc[Kk];
#pragma unroll
  for (int k = 0; k < Kk; ++k) acc[k] = 0.0f;
  for (int s = blk * srows; s < (blk + 1) * srows; ++s) {
    const float* xr = x + ((size_t)b * Ss + s) * Dd;
    for (int d = threadIdx.x; d < Dd; d += 256) {
      const float xv = xr[d];
#pragma unroll
      for (int k = 0; k < Kk; ++k) acc[k] += xv * router_w[k * Dd + d];
    }
  }
  __shared__ float red[256 * Kk];
#pragma unroll
  for (int k = 0; k < Kk; ++k) red[threadIdx.x * Kk + k] = acc[k];
  __syncthreads();
  for (int off = 128; off > 0; off >>= 1) {
    if (threadIdx.x < off) {
#pragma unroll
      for (int k = 0; k < Kk; ++k)
        red[threadIdx.x * Kk + k] += red[(threadIdx.x + off) * Kk + k];
    }
    __syncthreads();
  }
  if (threadIdx.x < Kk)
    part[(size_t)(b * RBLK + blk) * Kk + threadIdx.x] = red[threadIdx.x];
}

// Router pass 2: reduce partials, softmax -> weights[b][k]. One wave32.
__global__ __launch_bounds__(32) void router_finalize(const float* __restrict__ part,
                                                      const float* __restrict__ router_b,
                                                      float* __restrict__ weights) {
  const int tid = threadIdx.x;       // 0..31
  const int b = tid >> 3, k = tid & 7;
  float s = 0.0f;
  for (int i = 0; i < RBLK; ++i) s += part[(size_t)(b * RBLK + i) * Kk + k];
  const float logit = s / (float)Ss + router_b[k];
  __shared__ float l[32];
  l[tid] = logit;
  __syncthreads();
  float mx = l[b * 8];
#pragma unroll
  for (int kk = 1; kk < Kk; ++kk) mx = fmaxf(mx, l[b * 8 + kk]);
  const float e = __expf(logit - mx);
  __syncthreads();
  l[tid] = e;
  __syncthreads();
  float denom = 0.0f;
#pragma unroll
  for (int kk = 0; kk < Kk; ++kk) denom += l[b * 8 + kk];
  weights[b * Kk + k] = e / denom;
}

// Repack lora_B [K][D][R] fp32 -> Ball [D][KR] bf16 (Ball[o][k*16+r] = B_k[o][r])
__global__ __launch_bounds__(256) void prep_ball(const float* __restrict__ lora_B,
                                                 bf16_t* __restrict__ Ball) {
  const int idx = blockIdx.x * 256 + threadIdx.x;
  if (idx < Dd * KR) {
    const int o = idx / KR, kr = idx % KR;
    const int k = kr >> 4, r = kr & 15;
    Ball[idx] = (bf16_t)lora_B[((size_t)k * Dd + o) * Rr + r];
  }
}

// ---------------------------------------------------------------------------
// LoRA down-projection: hs[m][kr] = SCALING * w[b(m)][k] * sum_d x[m][d]*A[kr][d]
// M=8192, N=KR=128, K=4096. One block per 128 rows (BN covers all 128 cols).
// Register-pipelined staging: next tile loads overlap current tile's WMMAs.
// ---------------------------------------------------------------------------
__global__ __launch_bounds__(256) void lora_h_kernel(const float* __restrict__ x,
                                                     const float* __restrict__ lora_A,
                                                     const float* __restrict__ weights,
                                                     bf16_t* __restrict__ hs) {
  __shared__ bf16_t At[BM * LDT];
  __shared__ bf16_t Bt[BN * LDT];
  const int tid  = threadIdx.x;
  const int wave = tid >> 5;
  const int wm = (wave & 3) * 32;
  const int wn = (wave >> 2) * 64;
  const int m0 = blockIdx.x * BM;

  const int srow = tid >> 1;
  const int skh  = (tid & 1) * 32;
  const float* xp = x + (size_t)(m0 + srow) * Dd + skh;
  const float* ap = lora_A + (size_t)srow * Dd + skh;

  uint4 ra[4], rb[4];
  load_cvt32(xp, ra);
  load_cvt32(ap, rb);

  v8f acc[2][4];
#pragma unroll
  for (int i = 0; i < 2; ++i)
#pragma unroll
    for (int j = 0; j < 4; ++j) acc[i][j] = (v8f){0, 0, 0, 0, 0, 0, 0, 0};

  const int KT = Dd / BK;  // 64
  for (int kt = 0; kt < KT; ++kt) {
    __syncthreads();
    publish(At, srow, skh, ra);
    publish(Bt, srow, skh, rb);
    __syncthreads();
    if (kt + 1 < KT) {
      load_cvt32(xp + (kt + 1) * BK, ra);
      load_cvt32(ap + (kt + 1) * BK, rb);
    }
    if (kt + 2 < KT) {
      __builtin_prefetch(xp + (kt + 2) * BK, 0, 1);
      __builtin_prefetch(ap + (kt + 2) * BK, 0, 1);
    }
#pragma unroll
    for (int ks = 0; ks < 2; ++ks) {
      v16bf a0 = load_a_frag(At, wm, ks);
      v16bf a1 = load_a_frag(At, wm + 16, ks);
#pragma unroll
      for (int j = 0; j < 4; ++j) {
        v16bf bj = load_b_frag(Bt, wn + j * 16, ks);
        acc[0][j] = wmma_bf16(a0, bj, acc[0][j]);
        acc[1][j] = wmma_bf16(a1, bj, acc[1][j]);
      }
    }
  }

  // scale by router weight and SCALING, store bf16
  const int b = m0 / Ss;  // whole block lies in one batch (2048 % 128 == 0)
  const int lane = tid & 31;
  const int r = lane & 15, h = lane >> 4;
#pragma unroll
  for (int j = 0; j < 4; ++j) {
    const int gn = wn + j * 16 + r;
    const float w = SCALING * weights[b * Kk + (gn >> 4)];
#pragma unroll
    for (int i = 0; i < 2; ++i) {
#pragma unroll
      for (int v = 0; v < 8; ++v) {
        const int gm = m0 + wm + i * 16 + v + 8 * h;
        hs[(size_t)gm * KR + gn] = (bf16_t)(acc[i][j][v] * w);
      }
    }
  }
}

// ---------------------------------------------------------------------------
// Main fused GEMM: out = x @ W^T + bias + hs @ Ball^T
// M=8192, N=4096, K=4096 (+128 bf16 epilogue K-steps).
// ---------------------------------------------------------------------------
__global__ __launch_bounds__(256) void gemm_base_lora(const float* __restrict__ x,
                                                      const float* __restrict__ base_w,
                                                      const float* __restrict__ base_b,
                                                      const bf16_t* __restrict__ hs,
                                                      const bf16_t* __restrict__ Ball,
                                                      float* __restrict__ out) {
  __shared__ bf16_t At[BM * LDT];
  __shared__ bf16_t Bt[BN * LDT];
  const int tid  = threadIdx.x;
  const int wave = tid >> 5;
  const int wm = (wave & 3) * 32;
  const int wn = (wave >> 2) * 64;
  const int m0 = blockIdx.y * BM;
  const int n0 = blockIdx.x * BN;

  const int srow = tid >> 1;
  const int skh  = (tid & 1) * 32;
  const float* xp = x + (size_t)(m0 + srow) * Dd + skh;
  const float* wp = base_w + (size_t)(n0 + srow) * Dd + skh;

  uint4 ra[4], rb[4];
  load_cvt32(xp, ra);
  load_cvt32(wp, rb);

  v8f acc[2][4];
#pragma unroll
  for (int i = 0; i < 2; ++i)
#pragma unroll
    for (int j = 0; j < 4; ++j) acc[i][j] = (v8f){0, 0, 0, 0, 0, 0, 0, 0};

  const int KT = Dd / BK;  // 64
  for (int kt = 0; kt < KT; ++kt) {
    __syncthreads();
    publish(At, srow, skh, ra);
    publish(Bt, srow, skh, rb);
    __syncthreads();
    if (kt + 1 < KT) {
      load_cvt32(xp + (kt + 1) * BK, ra);
      load_cvt32(wp + (kt + 1) * BK, rb);
    }
    if (kt + 2 < KT) {
      __builtin_prefetch(xp + (kt + 2) * BK, 0, 1);
      __builtin_prefetch(wp + (kt + 2) * BK, 0, 1);
    }
#pragma unroll
    for (int ks = 0; ks < 2; ++ks) {
      v16bf a0 = load_a_frag(At, wm, ks);
      v16bf a1 = load_a_frag(At, wm + 16, ks);
#pragma unroll
      for (int j = 0; j < 4; ++j) {
        v16bf bj = load_b_frag(Bt, wn + j * 16, ks);
        acc[0][j] = wmma_bf16(a0, bj, acc[0][j]);
        acc[1][j] = wmma_bf16(a1, bj, acc[1][j]);
      }
    }
  }

  // rank-128 LoRA epilogue: 2 more 64-wide K-steps from precomputed bf16 operands
  const bf16_t* hp = hs + (size_t)(m0 + srow) * KR + skh;
  const bf16_t* bp = Ball + (size_t)(n0 + srow) * KR + skh;
#pragma unroll
  for (int kt = 0; kt < KR / BK; ++kt) {
    load_bf32(hp + kt * BK, ra);
    load_bf32(bp + kt * BK, rb);
    __syncthreads();
    publish(At, srow, skh, ra);
    publish(Bt, srow, skh, rb);
    __syncthreads();
#pragma unroll
    for (int ks = 0; ks < 2; ++ks) {
      v16bf a0 = load_a_frag(At, wm, ks);
      v16bf a1 = load_a_frag(At, wm + 16, ks);
#pragma unroll
      for (int j = 0; j < 4; ++j) {
        v16bf bj = load_b_frag(Bt, wn + j * 16, ks);
        acc[0][j] = wmma_bf16(a0, bj, acc[0][j]);
        acc[1][j] = wmma_bf16(a1, bj, acc[1][j]);
      }
    }
  }

  // Epilogue: + bias, write fp32. C/D layout: N = lane%16, M = vgpr + 8*(lane/16).
  const int lane = tid & 31;
  const int r = lane & 15, h = lane >> 4;
#pragma unroll
  for (int j = 0; j < 4; ++j) {
    const int gn = n0 + wn + j * 16 + r;
    const float bias = base_b[gn];
#pragma unroll
    for (int i = 0; i < 2; ++i) {
#pragma unroll
      for (int v = 0; v < 8; ++v) {
        const int gm = m0 + wm + i * 16 + v + 8 * h;
        out[(size_t)gm * Dd + gn] = acc[i][j][v] + bias;
      }
    }
  }
}

// ---------------------------------------------------------------------------
extern "C" void kernel_launch(void* const* d_in, const int* in_sizes, int n_in,
                              void* d_out, int out_size, void* d_ws, size_t ws_size,
                              hipStream_t stream) {
  const float* x        = (const float*)d_in[0];
  const float* base_w   = (const float*)d_in[1];
  const float* base_b   = (const float*)d_in[2];
  const float* lora_A   = (const float*)d_in[3];
  const float* lora_B   = (const float*)d_in[4];
  const float* router_w = (const float*)d_in[5];
  const float* router_b = (const float*)d_in[6];
  float* out = (float*)d_out;

  char* ws = (char*)d_ws;
  float*  part    = (float*)(ws);                       // 4*32*8 f32   = 4 KB
  float*  weights = (float*)(ws + 8192);                // 4*8 f32
  bf16_t* Ball    = (bf16_t*)(ws + 16384);              // 4096*128 bf16 = 1 MB
  bf16_t* hs      = (bf16_t*)(ws + 16384 + (size_t)Dd * KR * 2);  // 8192*128 bf16 = 2 MB

  router_partial<<<Bb * RBLK, 256, 0, stream>>>(x, router_w, part);
  router_finalize<<<1, 32, 0, stream>>>(part, router_b, weights);
  prep_ball<<<(Dd * KR + 255) / 256, 256, 0, stream>>>(lora_B, Ball);
  lora_h_kernel<<<M_TOTAL / BM, 256, 0, stream>>>(x, lora_A, weights, hs);

  dim3 grid(Dd / BN, M_TOTAL / BM);  // (32, 64)
  gemm_base_lora<<<grid, 256, 0, stream>>>(x, base_w, base_b, hs, Ball, out);
}
